// MultiHeadAttention_2010044695223
// MI455X (gfx1250) — matrix operand order repaired and checked by reading, not compiled, not run on hardware
//
#include <hip/hip_runtime.h>
#include <stdint.h>
#include <stddef.h>

// Problem constants (match reference)
#define Bsz  4
#define Sseq 2048
#define Dmod 1024
#define Hn   16
#define DKh  64

typedef __attribute__((ext_vector_type(16))) __bf16       v16bf;
typedef __attribute__((ext_vector_type(8)))  float        v8f;
typedef __attribute__((ext_vector_type(4)))  unsigned int u32x4;
typedef __attribute__((ext_vector_type(4)))  float        f32x4;
typedef __attribute__((ext_vector_type(4)))  __bf16       bf16x4;

union Frag {
    v16bf v;
    u32x4 q[2];
};

static __device__ __forceinline__ v8f wmma_bf16(const Frag& a, const Frag& b, v8f c) {
    // 8 args: (neg_a, A, neg_b, B, c_mod, C, reuse_a, reuse_b)
    return __builtin_amdgcn_wmma_f32_16x16x32_bf16(false, a.v, false, b.v, (short)0, c, false, false);
}

// ---------------------------------------------------------------------------
// Kernel 1: fp32 -> bf16 conversion, 4 elements/thread
// ---------------------------------------------------------------------------
__global__ __launch_bounds__(256)
void f2bf_kernel(const float* __restrict__ x, __bf16* __restrict__ y, int n4) {
    int i = blockIdx.x * 256 + threadIdx.x;
    if (i >= n4) return;
    f32x4 v = *(const f32x4*)(x + (size_t)i * 4);
    bf16x4 o;
    o.x = (__bf16)v.x; o.y = (__bf16)v.y; o.z = (__bf16)v.z; o.w = (__bf16)v.w;
    *(bf16x4*)(y + (size_t)i * 4) = o;
}

// ---------------------------------------------------------------------------
// Kernel 2: WMMA GEMM  C[M,N] = A[M,K] @ W[N,K]^T + bias
// Block: 256 threads = 8 waves; block tile 128(M) x 256(N); wave tile 64x64.
// Per K-step (32): 8 fragment loads feed 16 WMMAs -> ~32 FLOP/byte from cache.
// Fragments loaded straight from global (192MB L2) using the documented
// 16-bit A / B lane layouts (16B-aligned b128 loads).
// ---------------------------------------------------------------------------
__global__ __launch_bounds__(256)
void gemm_wmma_kernel(const __bf16* __restrict__ A, const __bf16* __restrict__ W,
                      const float* __restrict__ bias,
                      __bf16* __restrict__ Cb, float* __restrict__ Cf,
                      int M, int N, int K, int outF32)
{
    const int lane = threadIdx.x & 31;
    const int wid  = threadIdx.x >> 5;
    const int l16  = lane & 15;
    const int half = lane >> 4;

    const int nBlk = N / 256;
    const int bm   = blockIdx.x / nBlk;
    const int bn   = blockIdx.x % nBlk;
    const int Mb   = bm * 128 + (wid & 1) * 64;   // wave M base (4 x 16)
    const int Nb   = bn * 256 + (wid >> 1) * 64;  // wave N base (4 x 16)

    v8f acc[4][4];
    const v8f vzero = {0.f,0.f,0.f,0.f,0.f,0.f,0.f,0.f};
#pragma unroll
    for (int mi = 0; mi < 4; ++mi)
#pragma unroll
        for (int ni = 0; ni < 4; ++ni) acc[mi][ni] = vzero;

    for (int k0 = 0; k0 < K; k0 += 32) {
        Frag a[4], b[4];
#pragma unroll
        for (int mi = 0; mi < 4; ++mi) {
            // A 16x32 layout: lanes 0-15 K{0..7,16..23}, lanes 16-31 K{8..15,24..31}
            const __bf16* p = A + (size_t)(Mb + mi * 16 + l16) * K + k0 + half * 8;
            a[mi].q[0] = *(const u32x4*)p;
            a[mi].q[1] = *(const u32x4*)(p + 16);
        }
#pragma unroll
        for (int ni = 0; ni < 4; ++ni) {
            // B 32x16 layout: lane holds col n=l16, lanes 0-15 K{0..7,16..23}, lanes 16-31 K{8..15,24..31}
            const __bf16* p = W + (size_t)(Nb + ni * 16 + l16) * K + k0 + half * 8;
            b[ni].q[0] = *(const u32x4*)p;
            b[ni].q[1] = *(const u32x4*)(p + 16);
        }
#pragma unroll
        for (int mi = 0; mi < 4; ++mi)
#pragma unroll
            for (int ni = 0; ni < 4; ++ni)
                acc[mi][ni] = wmma_bf16(a[mi], b[ni], acc[mi][ni]);
    }

    // Epilogue: C layout lane=col n, VGPR j -> row m = half*8 + j
#pragma unroll
    for (int ni = 0; ni < 4; ++ni) {
        const int col = Nb + ni * 16 + l16;
        const float bv = bias[col];
#pragma unroll
        for (int mi = 0; mi < 4; ++mi) {
#pragma unroll
            for (int j = 0; j < 8; ++j) {
                const int row = Mb + mi * 16 + half * 8 + j;
                const float v = acc[mi][ni][j] + bv;
                if (outF32) Cf[(size_t)row * N + col] = v;
                else        Cb[(size_t)row * N + col] = (__bf16)v;
            }
        }
    }
}

// ---------------------------------------------------------------------------
// Kernel 3: transpose V [B,S,H,DK] -> Vt [B,H,DK,S]  (coalesced writes)
// ---------------------------------------------------------------------------
__global__ __launch_bounds__(256)
void transpose_v_kernel(const __bf16* __restrict__ Vp, __bf16* __restrict__ Vt) {
    size_t o = (size_t)blockIdx.x * 256 + threadIdx.x;
    const int s  = (int)(o & (Sseq - 1));
    size_t r = o >> 11;             // / Sseq
    const int dk = (int)(r & (DKh - 1));
    r >>= 6;                        // / DKh
    const int h = (int)(r & (Hn - 1));
    const int b = (int)(r >> 4);    // / Hn
    Vt[o] = Vp[((size_t)b * Sseq + s) * Dmod + h * DKh + dk];
}

// ---------------------------------------------------------------------------
// Kernel 4: flash attention. One wave owns 32 query rows (2 row-tiles) of one
// (b,h) so K/V fragments are reused across both tiles. Per 32-key block:
// 8 WMMAs (QK^T), online softmax with xor-shuffle row reductions, P re-layout
// via per-wave LDS, 8 WMMAs (P@V over 4 dk tiles using transposed V).
// ---------------------------------------------------------------------------
__global__ __launch_bounds__(256)
void attn_wmma_kernel(const __bf16* __restrict__ Q, const __bf16* __restrict__ Km,
                      const __bf16* __restrict__ Vt, __bf16* __restrict__ ctx)
{
    __shared__ __align__(16) __bf16 Pl[8][2][16 * 32];  // per-wave P staging (2 KB each)

    const int lane = threadIdx.x & 31;
    const int wid  = threadIdx.x >> 5;
    const int l16  = lane & 15;
    const int half = lane >> 4;

    const int gw = blockIdx.x * 8 + wid;      // global wave id, 4096 total
    const int qt = gw & 63;                   // S/32 = 64 q-tiles
    const int h  = (gw >> 6) & (Hn - 1);
    const int b  = gw >> 10;
    const int qb = qt * 32;

    const __bf16* qp = Q  + (size_t)b * Sseq * Dmod + h * DKh;
    const __bf16* kp = Km + (size_t)b * Sseq * Dmod + h * DKh;
    const __bf16* vt = Vt + ((size_t)b * Hn + h) * DKh * Sseq;

    // Q fragments: A layout, 2 row-tiles x dk chunks {0..31, 32..63}
    Frag qf[2][2];
#pragma unroll
    for (int t = 0; t < 2; ++t)
#pragma unroll
        for (int c = 0; c < 2; ++c) {
            const __bf16* p = qp + (size_t)(qb + t * 16 + l16) * Dmod + c * 32 + half * 8;
            qf[t][c].q[0] = *(const u32x4*)p;
            qf[t][c].q[1] = *(const u32x4*)(p + 16);
        }

    const v8f vzero = {0.f,0.f,0.f,0.f,0.f,0.f,0.f,0.f};
    float mr[2][8], lr[2][8];
    v8f acc[2][4];
#pragma unroll
    for (int t = 0; t < 2; ++t) {
#pragma unroll
        for (int j = 0; j < 8; ++j) { mr[t][j] = -1e30f; lr[t][j] = 0.f; }
#pragma unroll
        for (int d = 0; d < 4; ++d) acc[t][d] = vzero;
    }

    const float scale = 0.125f;  // 1/sqrt(64)

    for (int kb = 0; kb < Sseq; kb += 32) {
        // ---- K fragments (shared by both q row-tiles) ----
        Frag kf[2][2];
#pragma unroll
        for (int g = 0; g < 2; ++g)
#pragma unroll
            for (int c = 0; c < 2; ++c) {
                const __bf16* p = kp + (size_t)(kb + g * 16 + l16) * Dmod + c * 32 + half * 8;
                kf[g][c].q[0] = *(const u32x4*)p;
                kf[g][c].q[1] = *(const u32x4*)(p + 16);
            }

        // ---- scores: s[t][g] is 16q x 16keys ----
        v8f s[2][2];
#pragma unroll
        for (int t = 0; t < 2; ++t)
#pragma unroll
            for (int g = 0; g < 2; ++g) {
                v8f sv = vzero;
#pragma unroll
                for (int c = 0; c < 2; ++c) sv = wmma_bf16(qf[t][c], kf[g][c], sv);
                s[t][g] = sv;
            }

        // ---- online softmax per row-tile ----
#pragma unroll
        for (int t = 0; t < 2; ++t) {
            float mx[8];
#pragma unroll
            for (int j = 0; j < 8; ++j)
                mx[j] = fmaxf(s[t][0][j] * scale, s[t][1][j] * scale);
#pragma unroll
            for (int mask = 1; mask < 16; mask <<= 1)
#pragma unroll
                for (int j = 0; j < 8; ++j) mx[j] = fmaxf(mx[j], __shfl_xor(mx[j], mask, 32));

            float al[8];
#pragma unroll
            for (int j = 0; j < 8; ++j) {
                const float mn = fmaxf(mr[t][j], mx[j]);
                al[j] = __expf(mr[t][j] - mn);
                mr[t][j] = mn;
            }
            float rs[8];
#pragma unroll
            for (int j = 0; j < 8; ++j) {
                s[t][0][j] = __expf(s[t][0][j] * scale - mr[t][j]);
                s[t][1][j] = __expf(s[t][1][j] * scale - mr[t][j]);
                rs[j] = s[t][0][j] + s[t][1][j];
            }
#pragma unroll
            for (int mask = 1; mask < 16; mask <<= 1)
#pragma unroll
                for (int j = 0; j < 8; ++j) rs[j] += __shfl_xor(rs[j], mask, 32);
#pragma unroll
            for (int j = 0; j < 8; ++j) lr[t][j] = lr[t][j] * al[j] + rs[j];
#pragma unroll
            for (int d = 0; d < 4; ++d)
#pragma unroll
                for (int j = 0; j < 8; ++j) acc[t][d][j] *= al[j];
        }

        // ---- P (2 x 16x32) C-layout -> LDS -> A-fragment layout ----
#pragma unroll
        for (int t = 0; t < 2; ++t) {
            __bf16* pw = &Pl[wid][t][0];
#pragma unroll
            for (int g = 0; g < 2; ++g)
#pragma unroll
                for (int j = 0; j < 8; ++j)
                    pw[(half * 8 + j) * 32 + g * 16 + l16] = (__bf16)s[t][g][j];
        }

        asm volatile("s_wait_dscnt 0" ::: "memory");  // same-wave LDS is in-order

        Frag pf[2];
#pragma unroll
        for (int t = 0; t < 2; ++t) {
            const __bf16* pr = &Pl[wid][t][0] + l16 * 32 + half * 8;
            pf[t].q[0] = *(const u32x4*)pr;
            pf[t].q[1] = *(const u32x4*)(pr + 16);
        }

        // ---- ctx += P @ V : V fragments shared across both row-tiles ----
#pragma unroll
        for (int d = 0; d < 4; ++d) {
            Frag vf;
            const __bf16* p = vt + (size_t)(d * 16 + l16) * Sseq + kb + half * 8;
            vf.q[0] = *(const u32x4*)p;
            vf.q[1] = *(const u32x4*)(p + 16);
#pragma unroll
            for (int t = 0; t < 2; ++t) acc[t][d] = wmma_bf16(pf[t], vf, acc[t][d]);
        }
    }

    // ---- normalize and write ctx (bf16, [B,S,D] head slice) ----
    __bf16* cp = ctx + (size_t)b * Sseq * Dmod + h * DKh;
#pragma unroll
    for (int t = 0; t < 2; ++t)
#pragma unroll
        for (int d = 0; d < 4; ++d) {
            const int col = d * 16 + l16;
#pragma unroll
            for (int j = 0; j < 8; ++j) {
                const int row = qb + t * 16 + half * 8 + j;
                cp[(size_t)row * Dmod + col] = (__bf16)(acc[t][d][j] / lr[t][j]);
            }
        }
}

// ---------------------------------------------------------------------------
// Launch orchestration
// ---------------------------------------------------------------------------
extern "C" void kernel_launch(void* const* d_in, const int* in_sizes, int n_in,
                              void* d_out, int out_size, void* d_ws, size_t ws_size,
                              hipStream_t stream)
{
    (void)in_sizes; (void)n_in; (void)out_size; (void)ws_size;

    const float* query = (const float*)d_in[0];
    const float* key_  = (const float*)d_in[1];
    const float* value = (const float*)d_in[2];
    const float* Wq    = (const float*)d_in[3];
    const float* bq    = (const float*)d_in[4];
    const float* Wk    = (const float*)d_in[5];
    const float* bk    = (const float*)d_in[6];
    const float* Wv    = (const float*)d_in[7];
    const float* bv    = (const float*)d_in[8];
    const float* Wo    = (const float*)d_in[9];
    const float* bo    = (const float*)d_in[10];
    float* out = (float*)d_out;

    const size_t NX = (size_t)Bsz * Sseq * Dmod;  // 8388608 activation elements
    const size_t NW = (size_t)Dmod * Dmod;        // 1048576 weight elements

    char* p = (char*)d_ws;
    size_t off = 0;
    __bf16* qbf = (__bf16*)(p + off); off += NX * 2;
    __bf16* kbf = (__bf16*)(p + off); off += NX * 2;
    __bf16* vbf = (__bf16*)(p + off); off += NX * 2;
    __bf16* wqb = (__bf16*)(p + off); off += NW * 2;
    __bf16* wkb = (__bf16*)(p + off); off += NW * 2;
    __bf16* wvb = (__bf16*)(p + off); off += NW * 2;
    __bf16* wob = (__bf16*)(p + off); off += NW * 2;
    __bf16* Qp  = (__bf16*)(p + off); off += NX * 2;
    __bf16* Kp  = (__bf16*)(p + off); off += NX * 2;
    __bf16* Vp  = (__bf16*)(p + off); off += NX * 2;   // total 104 MB
    __bf16* Vt  = vbf;   // vbf dead after V projection
    __bf16* ctx = kbf;   // kbf dead after K projection

    const int nx4 = (int)(NX / 4), nw4 = (int)(NW / 4);
    const int gx = nx4 / 256, gwt = nw4 / 256;

    // 1) fp32 -> bf16
    f2bf_kernel<<<gx,  256, 0, stream>>>(query, qbf, nx4);
    f2bf_kernel<<<gx,  256, 0, stream>>>(key_,  kbf, nx4);
    f2bf_kernel<<<gx,  256, 0, stream>>>(value, vbf, nx4);
    f2bf_kernel<<<gwt, 256, 0, stream>>>(Wq, wqb, nw4);
    f2bf_kernel<<<gwt, 256, 0, stream>>>(Wk, wkb, nw4);
    f2bf_kernel<<<gwt, 256, 0, stream>>>(Wv, wvb, nw4);
    f2bf_kernel<<<gwt, 256, 0, stream>>>(Wo, wob, nw4);

    // 2) projections: [8192,1024] @ [1024,1024]^T + b  (bf16 out)
    const int M = Bsz * Sseq, N = Dmod, K = Dmod;
    const int gemmBlocks = (M / 128) * (N / 256);  // 256
    gemm_wmma_kernel<<<gemmBlocks, 256, 0, stream>>>(qbf, wqb, bq, Qp, nullptr, M, N, K, 0);
    gemm_wmma_kernel<<<gemmBlocks, 256, 0, stream>>>(kbf, wkb, bk, Kp, nullptr, M, N, K, 0);
    gemm_wmma_kernel<<<gemmBlocks, 256, 0, stream>>>(vbf, wvb, bv, Vp, nullptr, M, N, K, 0);

    // 3) V -> Vt [B,H,DK,S]
    transpose_v_kernel<<<(int)(NX / 256), 256, 0, stream>>>(Vp, Vt);

    // 4) attention: 4096 wave-tiles / 8 waves per block
    attn_wmma_kernel<<<512, 256, 0, stream>>>(Qp, Kp, Vt, ctx);

    // 5) output projection (fp32 out to d_out)
    gemm_wmma_kernel<<<gemmBlocks, 256, 0, stream>>>(ctx, wob, bo, nullptr, out, M, N, K, 1);
}
